// SimpleLSTM_39737037423257
// MI455X (gfx1250) — matrix operand order repaired and checked by reading, not compiled, not run on hardware
//
#include <hip/hip_runtime.h>

// ---------------- problem constants ----------------
constexpr int S    = 8;
constexpr int Bn   = 8192;
constexpr int INd  = 2;
constexpr int Ed   = 256;
constexpr int Hd   = 1024;
constexpr int OUTd = 2;
constexpr int PRED = 12;
constexpr int N4H  = 4 * Hd;      // 4096 gate rows in W_ih/W_hh

typedef __attribute__((ext_vector_type(16))) _Float16 v16h;
typedef __attribute__((ext_vector_type(8)))  _Float16 v8h;
typedef __attribute__((ext_vector_type(8)))  float    v8f;

// ---------------- utility kernels ----------------
__global__ void cvt_f32_to_f16(const float* __restrict__ src,
                               _Float16* __restrict__ dst, int n) {
    int i = blockIdx.x * blockDim.x + threadIdx.x;
    if (i < n) dst[i] = (_Float16)src[i];
}

__global__ void bias_combine(const float* __restrict__ a,
                             const float* __restrict__ b,
                             float* __restrict__ out, int n) {
    int i = blockIdx.x * blockDim.x + threadIdx.x;
    if (i < n) out[i] = a[i] + b[i];
}

// relu(x @ W_emb.T + b_emb), K=2, output f16.  rows*Ed threads.
__global__ void embed_relu_f16(const float* __restrict__ x,     // [rows, INd]
                               const float* __restrict__ Wemb,  // [Ed, INd]
                               const float* __restrict__ bemb,  // [Ed]
                               _Float16* __restrict__ out, int rows) {
    int i = blockIdx.x * blockDim.x + threadIdx.x;
    if (i >= rows * Ed) return;
    int r = i >> 8;          // / Ed (Ed == 256)
    int e = i & (Ed - 1);
    float v = fmaf(x[r * INd + 1], Wemb[e * INd + 1],
                   fmaf(x[r * INd + 0], Wemb[e * INd + 0], bemb[e]));
    out[i] = (_Float16)fmaxf(v, 0.0f);
}

// ---------------- WMMA fragment loads (CDNA5 16x16x32 f16 layouts) ----------------
// A (16x32, f16): lane m = L&15; lanes 0-15 hold K=0..7 (v0-3) & 16..23 (v4-7),
// lanes 16-31 hold K=8..15 & 24..31  ->  8 halfs at k0+8*grp, 8 halfs at +16.
__device__ __forceinline__ v16h load_a_frag(const _Float16* __restrict__ X,
                                            int K, int m0, int k0, int lane) {
    const _Float16* p = X + (size_t)(m0 + (lane & 15)) * K + k0 + ((lane >> 4) << 3);
    v8h lo = *(const v8h*)p;          // global_load_b128
    v8h hi = *(const v8h*)(p + 16);   // global_load_b128
    v16h a;
#pragma unroll
    for (int i = 0; i < 8; ++i) { a[i] = lo[i]; a[i + 8] = hi[i]; }
    return a;
}

// B (32x16, f16): lane n = L&15; lanes 0-15 hold K=0..7 & 16..23, lanes 16-31 K=8..15 & 24..31
// (same k order as A) -> 8 halfs at k0+8*grp, 8 halfs at +16 of weight row n (B[k][n] = W[n][k]).
__device__ __forceinline__ v16h load_b_frag(const _Float16* __restrict__ W,
                                            int K, int n0, int k0, int lane) {
    const _Float16* p = W + (size_t)(n0 + (lane & 15)) * K + k0 + ((lane >> 4) << 3);
    v8h lo = *(const v8h*)p;          // global_load_b128
    v8h hi = *(const v8h*)(p + 16);   // global_load_b128
    v16h b;
#pragma unroll
    for (int i = 0; i < 8; ++i) { b[i] = lo[i]; b[i + 8] = hi[i]; }
    return b;
}

__device__ __forceinline__ float sigm(float x) {
    return 1.0f / (1.0f + __expf(-x));
}

// ---------------- fully fused LSTM cell ----------------
// For a 128-row batch tile and 32-col hidden tile, accumulate all four gates
// (weight rows g*Hd + j) with WMMA, then apply bias + activations + c/h update
// entirely in registers.  No gate buffer ever touches HBM.
//   grid: (Hd/32, Bn/128), block: 256 (8 waves)
//   wave tile: 32 rows x 16 hidden cols x 4 gates  (8 wmma tiles, 64 acc VGPRs)
__global__ __launch_bounds__(256)
void lstm_cell_fused(const _Float16* __restrict__ X,  int K1,
                     const _Float16* __restrict__ Wih,
                     const _Float16* __restrict__ Hp, int K2,
                     const _Float16* __restrict__ Whh,
                     const float* __restrict__ bias,   // combined [4H]
                     float* __restrict__ c,            // [Bn, Hd] in/out
                     _Float16* __restrict__ hout) {    // [Bn, Hd] out (f16)
    const int lane = threadIdx.x & 31;
    const int w    = threadIdx.x >> 5;      // 0..7
    const int wm   = w >> 1;                // 0..3  (M dir)
    const int wn   = w & 1;                 // 0..1  (hidden-col dir)
    const int m0   = blockIdx.y * 128 + wm * 32;
    const int j0   = blockIdx.x * 32 + wn * 16;   // hidden column base

    v8f acc[4][2];                                // [gate][m-subtile]
#pragma unroll
    for (int g = 0; g < 4; ++g)
#pragma unroll
        for (int mi = 0; mi < 2; ++mi) acc[g][mi] = (v8f)(0.0f);

    // ---- X @ Wih.T (gate rows g*Hd + j) ----
    for (int k0 = 0; k0 < K1; k0 += 32) {
        v16h a0 = load_a_frag(X, K1, m0,      k0, lane);
        v16h a1 = load_a_frag(X, K1, m0 + 16, k0, lane);
        if (k0 + 32 < K1)   // prefetch next weight K-tile (global_prefetch_b8)
            __builtin_prefetch(Wih + (size_t)(j0 + (lane & 15)) * K1 + k0 + 32, 0, 1);
#pragma unroll
        for (int g = 0; g < 4; ++g) {
            v16h b = load_b_frag(Wih, K1, g * Hd + j0, k0, lane);
            acc[g][0] = __builtin_amdgcn_wmma_f32_16x16x32_f16(
                false, a0, false, b, (short)0, acc[g][0], false, false);
            acc[g][1] = __builtin_amdgcn_wmma_f32_16x16x32_f16(
                false, a1, false, b, (short)0, acc[g][1], false, false);
        }
    }
    // ---- Hp @ Whh.T ----
    for (int k0 = 0; k0 < K2; k0 += 32) {
        v16h a0 = load_a_frag(Hp, K2, m0,      k0, lane);
        v16h a1 = load_a_frag(Hp, K2, m0 + 16, k0, lane);
        if (k0 + 32 < K2)
            __builtin_prefetch(Whh + (size_t)(j0 + (lane & 15)) * K2 + k0 + 32, 0, 1);
#pragma unroll
        for (int g = 0; g < 4; ++g) {
            v16h b = load_b_frag(Whh, K2, g * Hd + j0, k0, lane);
            acc[g][0] = __builtin_amdgcn_wmma_f32_16x16x32_f16(
                false, a0, false, b, (short)0, acc[g][0], false, false);
            acc[g][1] = __builtin_amdgcn_wmma_f32_16x16x32_f16(
                false, a1, false, b, (short)0, acc[g][1], false, false);
        }
    }

    // ---- epilogue: bias + activations + state update, all in registers ----
    // C/D layout: lane col = L&15; VGPR v -> row = v + 8*(L>>4)
    const int col   = j0 + (lane & 15);
    const int rbase = m0 + ((lane >> 4) << 3);
    const float bi = bias[0 * Hd + col];
    const float bf = bias[1 * Hd + col];
    const float bg = bias[2 * Hd + col];
    const float bo = bias[3 * Hd + col];
#pragma unroll
    for (int mi = 0; mi < 2; ++mi) {
#pragma unroll
        for (int v = 0; v < 8; ++v) {
            int row = rbase + mi * 16 + v;
            size_t idx = (size_t)row * Hd + col;
            float gi = acc[0][mi][v] + bi;
            float gf = acc[1][mi][v] + bf;
            float gg = acc[2][mi][v] + bg;
            float go = acc[3][mi][v] + bo;
            float cv = sigm(gf) * c[idx] + sigm(gi) * tanhf(gg);
            c[idx]    = cv;
            hout[idx] = (_Float16)(sigm(go) * tanhf(cv));
        }
    }
}

// ---------------- decode head: MLP + displacement, thread per batch row ----------------
__global__ __launch_bounds__(256)
void mlp_decode(const _Float16* __restrict__ h1,  // [Bn, H]
                const float* __restrict__ W1, const float* __restrict__ b1,   // [32,H],[32]
                const float* __restrict__ W2, const float* __restrict__ b2,   // [16,32],[16]
                const float* __restrict__ W3, const float* __restrict__ b3,   // [2,16],[2]
                float* __restrict__ curr,          // [Bn,2] in/out
                float* __restrict__ pred) {        // [Bn,2] out
    int b = blockIdx.x * blockDim.x + threadIdx.x;
    if (b >= Bn) return;
    float a1[32];
#pragma unroll
    for (int j = 0; j < 32; ++j) a1[j] = b1[j];
    const _Float16* hr = h1 + (size_t)b * Hd;
    for (int k = 0; k < Hd; ++k) {
        float x = (float)hr[k];
#pragma unroll
        for (int j = 0; j < 32; ++j) a1[j] = fmaf(x, W1[j * Hd + k], a1[j]);
    }
#pragma unroll
    for (int j = 0; j < 32; ++j) a1[j] = fmaxf(a1[j], 0.0f);

    float a2[16];
#pragma unroll
    for (int j = 0; j < 16; ++j) a2[j] = b2[j];
#pragma unroll
    for (int k = 0; k < 32; ++k) {
        float x = a1[k];
#pragma unroll
        for (int j = 0; j < 16; ++j) a2[j] = fmaf(x, W2[j * 32 + k], a2[j]);
    }
#pragma unroll
    for (int j = 0; j < 16; ++j) a2[j] = fmaxf(a2[j], 0.0f);

    float o0 = b3[0], o1 = b3[1];
#pragma unroll
    for (int k = 0; k < 16; ++k) {
        o0 = fmaf(a2[k], W3[k],      o0);
        o1 = fmaf(a2[k], W3[16 + k], o1);
    }
    float n0 = curr[b * 2 + 0] + o0;
    float n1 = curr[b * 2 + 1] + o1;
    curr[b * 2 + 0] = n0;  curr[b * 2 + 1] = n1;
    pred[b * 2 + 0] = n0;  pred[b * 2 + 1] = n1;
}

// ---------------- host orchestration ----------------
static inline int ceil_div(int a, int b) { return (a + b - 1) / b; }

extern "C" void kernel_launch(void* const* d_in, const int* in_sizes, int n_in,
                              void* d_out, int out_size, void* d_ws, size_t ws_size,
                              hipStream_t stream) {
    (void)in_sizes; (void)n_in; (void)out_size; (void)ws_size;
    // inputs (setup_inputs order)
    const float* obs   = (const float*)d_in[0];   // [S, B, IN]
    const float* Wemb  = (const float*)d_in[1];   // [E, IN]
    const float* bemb  = (const float*)d_in[2];
    const float* Wih0  = (const float*)d_in[3];   // [4H, E]
    const float* Whh0  = (const float*)d_in[4];   // [4H, H]
    const float* bih0  = (const float*)d_in[5];
    const float* bhh0  = (const float*)d_in[6];
    const float* Wih1  = (const float*)d_in[7];   // [4H, H]
    const float* Whh1  = (const float*)d_in[8];
    const float* bih1  = (const float*)d_in[9];
    const float* bhh1  = (const float*)d_in[10];
    const float* W1    = (const float*)d_in[11];
    const float* b1    = (const float*)d_in[12];
    const float* W2    = (const float*)d_in[13];
    const float* b2    = (const float*)d_in[14];
    const float* W3    = (const float*)d_in[15];
    const float* b3    = (const float*)d_in[16];
    float* preds = (float*)d_out;                 // [PRED, B, OUT]

    // workspace carve-out (256B aligned)
    size_t off = 0;
    auto carve = [&](size_t bytes) -> void* {
        void* p = (char*)d_ws + off;
        off += (bytes + 255) & ~(size_t)255;
        return p;
    };
    _Float16* wih0h  = (_Float16*)carve((size_t)N4H * Ed * 2);
    _Float16* whh0h  = (_Float16*)carve((size_t)N4H * Hd * 2);
    _Float16* wih1h  = (_Float16*)carve((size_t)N4H * Hd * 2);
    _Float16* whh1h  = (_Float16*)carve((size_t)N4H * Hd * 2);
    float*    bias0  = (float*)   carve((size_t)N4H * 4);
    float*    bias1  = (float*)   carve((size_t)N4H * 4);
    _Float16* embobs = (_Float16*)carve((size_t)S * Bn * Ed * 2);
    _Float16* embcur = (_Float16*)carve((size_t)Bn * Ed * 2);
    _Float16* h0f    = (_Float16*)carve((size_t)Bn * Hd * 2);
    _Float16* h1f    = (_Float16*)carve((size_t)Bn * Hd * 2);
    float*    c0     = (float*)   carve((size_t)Bn * Hd * 4);
    float*    c1     = (float*)   carve((size_t)Bn * Hd * 4);
    float*    curr   = (float*)   carve((size_t)Bn * OUTd * 4);

    const int TB = 256;
    // weight precision conversion (weights live in L2 across the whole run)
    cvt_f32_to_f16<<<ceil_div(N4H * Ed, TB), TB, 0, stream>>>(Wih0, wih0h, N4H * Ed);
    cvt_f32_to_f16<<<ceil_div(N4H * Hd, TB), TB, 0, stream>>>(Whh0, whh0h, N4H * Hd);
    cvt_f32_to_f16<<<ceil_div(N4H * Hd, TB), TB, 0, stream>>>(Wih1, wih1h, N4H * Hd);
    cvt_f32_to_f16<<<ceil_div(N4H * Hd, TB), TB, 0, stream>>>(Whh1, whh1h, N4H * Hd);
    bias_combine<<<ceil_div(N4H, TB), TB, 0, stream>>>(bih0, bhh0, bias0, N4H);
    bias_combine<<<ceil_div(N4H, TB), TB, 0, stream>>>(bih1, bhh1, bias1, N4H);

    // init recurrent state to zero
    hipMemsetAsync(h0f, 0, (size_t)Bn * Hd * 2, stream);
    hipMemsetAsync(h1f, 0, (size_t)Bn * Hd * 2, stream);
    hipMemsetAsync(c0,  0, (size_t)Bn * Hd * 4, stream);
    hipMemsetAsync(c1,  0, (size_t)Bn * Hd * 4, stream);

    // embed all observed steps at once
    embed_relu_f16<<<ceil_div(S * Bn * Ed, TB), TB, 0, stream>>>(obs, Wemb, bemb,
                                                                embobs, S * Bn);

    const dim3 cgrid(Hd / 32, Bn / 128);     // 32 x 64 workgroups

    // ---- encoder: 8 steps ----
    for (int s = 0; s < S; ++s) {
        const _Float16* xs = embobs + (size_t)s * Bn * Ed;
        lstm_cell_fused<<<cgrid, TB, 0, stream>>>(xs, Ed, wih0h, h0f, Hd, whh0h,
                                                  bias0, c0, h0f);
        lstm_cell_fused<<<cgrid, TB, 0, stream>>>(h0f, Hd, wih1h, h1f, Hd, whh1h,
                                                  bias1, c1, h1f);
    }

    // ---- decoder: 12 autoregressive steps ----
    hipMemcpyAsync(curr, obs + (size_t)(S - 1) * Bn * INd, (size_t)Bn * INd * 4,
                   hipMemcpyDeviceToDevice, stream);
    embed_relu_f16<<<ceil_div(Bn * Ed, TB), TB, 0, stream>>>(curr, Wemb, bemb,
                                                             embcur, Bn);
    for (int p = 0; p < PRED; ++p) {
        lstm_cell_fused<<<cgrid, TB, 0, stream>>>(embcur, Ed, wih0h, h0f, Hd, whh0h,
                                                  bias0, c0, h0f);
        lstm_cell_fused<<<cgrid, TB, 0, stream>>>(h0f, Hd, wih1h, h1f, Hd, whh1h,
                                                  bias1, c1, h1f);
        mlp_decode<<<ceil_div(Bn, TB), TB, 0, stream>>>(h1f, W1, b1, W2, b2, W3, b3,
                                                        curr,
                                                        preds + (size_t)p * Bn * OUTd);
        embed_relu_f16<<<ceil_div(Bn * Ed, TB), TB, 0, stream>>>(curr, Wemb, bemb,
                                                                 embcur, Bn);
    }
}
